// OCRMemoryLayer_72584947303021
// MI455X (gfx1250) — matrix operand order repaired and checked by reading, not compiled, not run on hardware
//
#include <hip/hip_runtime.h>
#include <math.h>

// ---------------------------------------------------------------------------
// OCR memory layer for MI455X (gfx1250, wave32, WMMA).
// All heavy math runs through v_wmma_f32_16x16x32_bf16 (fp32 accumulate).
// ocr_bias is a per-(b,h) additive constant on pre-softmax scores; softmax is
// shift-invariant per row, so it cancels exactly and is intentionally unused.
// ---------------------------------------------------------------------------

typedef __attribute__((ext_vector_type(16))) __bf16 v16bf;
typedef __attribute__((ext_vector_type(2)))  __bf16 v2bf;
typedef __attribute__((ext_vector_type(8)))  float  v8f;

union FragU { uint4 q[2]; v16bf v; };

// Pack two fp32 -> packed bf16x2.  Prefer the packed-cvt builtin when the
// toolchain declares it; otherwise cast through native __bf16 so the backend
// picks the hardware conversion (and packs without sub-register moves).
__device__ __forceinline__ unsigned int pack2(float a, float b) {
#if __has_builtin(__builtin_amdgcn_cvt_pk_bf16_f32)
    union { v2bf v; unsigned int u; } cv;
    cv.v = __builtin_amdgcn_cvt_pk_bf16_f32(a, b);
    return cv.u;
#else
    union { v2bf v; unsigned int u; } cv;
    cv.v.x = (__bf16)a;
    cv.v.y = (__bf16)b;
    return cv.u;
#endif
}

// A-fragment (16x32 bf16): lanes 0-15 hold K=0..7 (VGPR0-3) and K=16..23
// (VGPR4-7); lanes 16-31 hold K=8..15 and K=24..31.  LDS tile row-major.
__device__ __forceinline__ v16bf frag_a(const unsigned short* ls, int row,
                                        int stride, int k0, int lane) {
    int kh = ((lane >> 4) & 1) * 8;
    const unsigned short* p = ls + row * stride + k0 + kh;
    FragU f;
    f.q[0] = *(const uint4*)p;        // K kh..kh+7
    f.q[1] = *(const uint4*)(p + 16); // K kh+16..kh+23
    return f.v;
}
// B-fragment (32x16 bf16): lane = N column; lanes 0-15 hold K=0..7 and
// K=16..23; lanes 16-31 hold K=8..15 and K=24..31.  LDS tile is column-major (B^T row-major).
__device__ __forceinline__ v16bf frag_b(const unsigned short* ls, int col,
                                        int stride, int k0, int lane) {
    int kh = ((lane >> 4) & 1) * 8;
    const unsigned short* p = ls + col * stride + k0 + kh;
    FragU f;
    f.q[0] = *(const uint4*)p;        // K kh..kh+7
    f.q[1] = *(const uint4*)(p + 16); // K kh+16..kh+23
    return f.v;
}
__device__ __forceinline__ v8f wmma_bf16(v16bf a, v16bf b, v8f c) {
    return __builtin_amdgcn_wmma_f32_16x16x32_bf16(false, a, false, b,
                                                   (short)0, c, false, false);
}

// ---------------------------------------------------------------------------
// Tiled GEMM: C[MxN] = act(A[MxK] @ B[KxN] + bias), A optionally split in K
// (virtual concat: cols [0,ksplit) from A, [ksplit,K) from A2).
// Block = 256 threads = 8 waves; tile 128x128, BK=32; each wave owns 32x64.
// Double-buffered LDS; global loads for tile k+1 issued before compute of
// tile k, converted+stored after the WMMA chain; one barrier per K-tile.
// ---------------------------------------------------------------------------
__global__ __launch_bounds__(256) void gemm_bf16_kernel(
    const float* __restrict__ A,  int lda,
    const float* __restrict__ A2, int lda2, int ksplit,
    const float* __restrict__ Bw, int ldb,
    const float* __restrict__ bias,
    float* __restrict__ C, int ldc,
    int K, int act)
{
    __shared__ unsigned short sA[2][128 * 32];   // row-major, stride 32
    __shared__ unsigned short sB[2][128 * 32];   // B^T: [n][k], stride 32

    const int tid = threadIdx.x, lane = tid & 31, w = tid >> 5;
    const int wr = w >> 1, wc = w & 1, ln = lane & 15, hh = lane >> 4;
    const int bm0 = blockIdx.y * 128, bn0 = blockIdx.x * 128;

    v8f acc[2][4];
#pragma unroll
    for (int i = 0; i < 2; ++i)
#pragma unroll
        for (int j = 0; j < 4; ++j)
#pragma unroll
            for (int e = 0; e < 8; ++e) acc[i][j][e] = 0.f;

    const int ar = tid >> 1, ak = (tid & 1) * 16;       // A staging: 16 floats
    const int bn = tid & 127, bkg = (tid >> 7) * 16;    // B staging: 16 floats

    float4 ra[4];
    float  rb[16];

    auto loadA = [&](int k0) {
        int kc = k0 + ak;
        const float* src; int ld;
        if (kc < ksplit) { src = A;  ld = lda;  }
        else             { src = A2; ld = lda2; kc -= ksplit; }
        const float* p = src + (size_t)((bm0 + ar) * ld + kc);
        ra[0] = ((const float4*)p)[0];
        ra[1] = ((const float4*)p)[1];
        ra[2] = ((const float4*)p)[2];
        ra[3] = ((const float4*)p)[3];
        if (k0 + 32 < K) {  // prefetch tile k+2 (global_prefetch_b8)
            int kn = k0 + 32 + ak;
            const float* ps; int ldn;
            if (kn < ksplit) { ps = A;  ldn = lda;  }
            else             { ps = A2; ldn = lda2; kn -= ksplit; }
            __builtin_prefetch(ps + (size_t)((bm0 + ar) * ldn + kn), 0, 0);
        }
    };
    auto loadB = [&](int k0) {
        const float* p = Bw + (size_t)((k0 + bkg) * ldb + bn0 + bn);
#pragma unroll
        for (int i = 0; i < 16; ++i) { rb[i] = *p; p += ldb; }
    };
    auto storeA = [&](int buf) {
        uint4* dst = (uint4*)&sA[buf][ar * 32 + ak];
        dst[0] = make_uint4(pack2(ra[0].x, ra[0].y), pack2(ra[0].z, ra[0].w),
                            pack2(ra[1].x, ra[1].y), pack2(ra[1].z, ra[1].w));
        dst[1] = make_uint4(pack2(ra[2].x, ra[2].y), pack2(ra[2].z, ra[2].w),
                            pack2(ra[3].x, ra[3].y), pack2(ra[3].z, ra[3].w));
    };
    auto storeB = [&](int buf) {
        uint4* dst = (uint4*)&sB[buf][bn * 32 + bkg];
        dst[0] = make_uint4(pack2(rb[0],  rb[1]),  pack2(rb[2],  rb[3]),
                            pack2(rb[4],  rb[5]),  pack2(rb[6],  rb[7]));
        dst[1] = make_uint4(pack2(rb[8],  rb[9]),  pack2(rb[10], rb[11]),
                            pack2(rb[12], rb[13]), pack2(rb[14], rb[15]));
    };

    loadA(0); loadB(0);
    storeA(0); storeB(0);
    __syncthreads();

    const int nk = K / 32;
    for (int kt = 0; kt < nk; ++kt) {
        const int cur = kt & 1;
        if (kt + 1 < nk) { loadA((kt + 1) * 32); loadB((kt + 1) * 32); }

        // ---- all 12 ds_load_b128 up front, then the 8-WMMA chain ----
        const unsigned short* pA = sA[cur];
        const unsigned short* pB = sB[cur];
        v16bf a0 = frag_a(pA, wr * 32 + ln,      32, 0, lane);
        v16bf a1 = frag_a(pA, wr * 32 + 16 + ln, 32, 0, lane);
        v16bf b0 = frag_b(pB, wc * 64 +  0 + ln, 32, 0, lane);
        v16bf b1 = frag_b(pB, wc * 64 + 16 + ln, 32, 0, lane);
        v16bf b2 = frag_b(pB, wc * 64 + 32 + ln, 32, 0, lane);
        v16bf b3 = frag_b(pB, wc * 64 + 48 + ln, 32, 0, lane);
        acc[0][0] = wmma_bf16(a0, b0, acc[0][0]);
        acc[1][0] = wmma_bf16(a1, b0, acc[1][0]);
        acc[0][1] = wmma_bf16(a0, b1, acc[0][1]);
        acc[1][1] = wmma_bf16(a1, b1, acc[1][1]);
        acc[0][2] = wmma_bf16(a0, b2, acc[0][2]);
        acc[1][2] = wmma_bf16(a1, b2, acc[1][2]);
        acc[0][3] = wmma_bf16(a0, b3, acc[0][3]);
        acc[1][3] = wmma_bf16(a1, b3, acc[1][3]);

        if (kt + 1 < nk) { storeA(cur ^ 1); storeB(cur ^ 1); }
        __syncthreads();
    }

    // ---- epilogue: bias (+ exact GELU) ----
#pragma unroll
    for (int ni = 0; ni < 4; ++ni) {
        int gcol = bn0 + wc * 64 + ni * 16 + ln;
        float bv = bias ? bias[gcol] : 0.f;
#pragma unroll
        for (int mi = 0; mi < 2; ++mi) {
#pragma unroll
            for (int e = 0; e < 8; ++e) {
                int grow = bm0 + wr * 32 + mi * 16 + e + hh * 8;
                float val = acc[mi][ni][e] + bv;
                if (act == 1)
                    val = 0.5f * val * (1.0f + erff(val * 0.70710678118f));
                C[(size_t)(grow * ldc) + gcol] = val;
            }
        }
    }
}

// ---------------------------------------------------------------------------
// Fused flash attention: per (b,h) and 64-query tile; 128 threads = 4 waves,
// each wave owns 16 query rows.  Online softmax; O accumulated in WMMA regs.
// S=2048, M=512, H=8, DK=64.  Q pre-scaled by 1/sqrt(64).
// ---------------------------------------------------------------------------
__global__ __launch_bounds__(128) void attn_kernel(
    const float* __restrict__ Qg, const float* __restrict__ Kg,
    const float* __restrict__ Vg, float* __restrict__ Og)
{
    __shared__ unsigned short sQ[64 * 64];   // [q][d]   bf16
    __shared__ unsigned short sK[64 * 64];   // [key][d] bf16 (== B^T for QK^T)
    __shared__ unsigned short sVT[64 * 64];  // [d][key] bf16 (== B^T for PV)
    __shared__ unsigned short sP[64 * 64];   // [q][key] bf16 probabilities
    __shared__ float sS[64 * 64];            // fp32 scores
    __shared__ float rmax[64], rsum[64], rscale[64];

    const int tid = threadIdx.x, lane = tid & 31, w = tid >> 5;
    const int ln = lane & 15, hh = lane >> 4;
    const int qb = blockIdx.x * 64;
    const int h = blockIdx.y & 7, b = blockIdx.y >> 3;
    const size_t rowbase = (size_t)b * 2048;
    const int hoff = h * 64;

    // ---- stage Q (scaled by 1/sqrt(DK)) ----
    {
        int r = tid >> 1, half = (tid & 1) * 32;
        const float* p = Qg + (rowbase + qb + r) * 512 + hoff + half;
        unsigned int pk[16];
#pragma unroll
        for (int i = 0; i < 8; ++i) {
            float4 f = ((const float4*)p)[i];
            pk[2 * i]     = pack2(f.x * 0.125f, f.y * 0.125f);
            pk[2 * i + 1] = pack2(f.z * 0.125f, f.w * 0.125f);
        }
        uint4* dst = (uint4*)&sQ[r * 64 + half];
#pragma unroll
        for (int i = 0; i < 4; ++i)
            dst[i] = make_uint4(pk[4*i], pk[4*i+1], pk[4*i+2], pk[4*i+3]);
    }
    if (tid < 64) { rmax[tid] = -INFINITY; rsum[tid] = 0.f; }
    v8f oacc[4];
#pragma unroll
    for (int ni = 0; ni < 4; ++ni)
#pragma unroll
        for (int e = 0; e < 8; ++e) oacc[ni][e] = 0.f;
    __syncthreads();

    for (int kb = 0; kb < 2048; kb += 64) {
        // ---- stage K tile ----
        {
            int r = tid >> 1, half = (tid & 1) * 32;
            const float* p = Kg + (rowbase + kb + r) * 512 + hoff + half;
            unsigned int pk[16];
#pragma unroll
            for (int i = 0; i < 8; ++i) {
                float4 f = ((const float4*)p)[i];
                pk[2 * i]     = pack2(f.x, f.y);
                pk[2 * i + 1] = pack2(f.z, f.w);
            }
            uint4* dst = (uint4*)&sK[r * 64 + half];
#pragma unroll
            for (int i = 0; i < 4; ++i)
                dst[i] = make_uint4(pk[4*i], pk[4*i+1], pk[4*i+2], pk[4*i+3]);
        }
        // ---- stage V transposed ----
        {
            int d = tid & 63, kg2 = (tid >> 6) * 32;
            unsigned int pk[16];
#pragma unroll
            for (int i = 0; i < 16; ++i) {
                float a = Vg[(rowbase + kb + kg2 + 2*i    ) * 512 + hoff + d];
                float c = Vg[(rowbase + kb + kg2 + 2*i + 1) * 512 + hoff + d];
                pk[i] = pack2(a, c);
            }
            uint4* dst = (uint4*)&sVT[d * 64 + kg2];
#pragma unroll
            for (int i = 0; i < 4; ++i)
                dst[i] = make_uint4(pk[4*i], pk[4*i+1], pk[4*i+2], pk[4*i+3]);
        }
        __syncthreads();

        // ---- S = Qs @ K^T (each wave: 16 q-rows x 64 keys) ----
        v8f sacc[4];
#pragma unroll
        for (int ni = 0; ni < 4; ++ni)
#pragma unroll
            for (int e = 0; e < 8; ++e) sacc[ni][e] = 0.f;
#pragma unroll
        for (int k0 = 0; k0 < 64; k0 += 32) {
            v16bf aq = frag_a(sQ, w * 16 + ln, 64, k0, lane);
            v16bf b0 = frag_b(sK,  0 + ln, 64, k0, lane);
            v16bf b1 = frag_b(sK, 16 + ln, 64, k0, lane);
            v16bf b2 = frag_b(sK, 32 + ln, 64, k0, lane);
            v16bf b3 = frag_b(sK, 48 + ln, 64, k0, lane);
            sacc[0] = wmma_bf16(aq, b0, sacc[0]);
            sacc[1] = wmma_bf16(aq, b1, sacc[1]);
            sacc[2] = wmma_bf16(aq, b2, sacc[2]);
            sacc[3] = wmma_bf16(aq, b3, sacc[3]);
        }
#pragma unroll
        for (int ni = 0; ni < 4; ++ni)
#pragma unroll
            for (int e = 0; e < 8; ++e)
                sS[(w * 16 + e + hh * 8) * 64 + ni * 16 + ln] = sacc[ni][e];
        __syncthreads();

        // ---- online softmax (one thread per query row) ----
        if (tid < 64) {
            int r = tid;
            const float* srow = &sS[r * 64];
            float mo = rmax[r], mn = mo;
#pragma unroll 8
            for (int j = 0; j < 64; ++j) mn = fmaxf(mn, srow[j]);
            float rs = __expf(mo - mn);            // 0 on first tile
            float s = rsum[r] * rs;
            unsigned int* prow = (unsigned int*)&sP[r * 64];
#pragma unroll 8
            for (int j = 0; j < 32; ++j) {
                float p0 = __expf(srow[2*j]     - mn);
                float p1 = __expf(srow[2*j + 1] - mn);
                s += p0 + p1;
                prow[j] = pack2(p0, p1);
            }
            rmax[r] = mn; rsum[r] = s; rscale[r] = rs;
        }
        __syncthreads();

        // ---- rescale O, then O += P @ V ----
        float sc[8];
#pragma unroll
        for (int e = 0; e < 8; ++e) sc[e] = rscale[w * 16 + e + hh * 8];
#pragma unroll
        for (int ni = 0; ni < 4; ++ni)
#pragma unroll
            for (int e = 0; e < 8; ++e) oacc[ni][e] *= sc[e];
#pragma unroll
        for (int k0 = 0; k0 < 64; k0 += 32) {
            v16bf ap = frag_a(sP, w * 16 + ln, 64, k0, lane);
            v16bf b0 = frag_b(sVT,  0 + ln, 64, k0, lane);
            v16bf b1 = frag_b(sVT, 16 + ln, 64, k0, lane);
            v16bf b2 = frag_b(sVT, 32 + ln, 64, k0, lane);
            v16bf b3 = frag_b(sVT, 48 + ln, 64, k0, lane);
            oacc[0] = wmma_bf16(ap, b0, oacc[0]);
            oacc[1] = wmma_bf16(ap, b1, oacc[1]);
            oacc[2] = wmma_bf16(ap, b2, oacc[2]);
            oacc[3] = wmma_bf16(ap, b3, oacc[3]);
        }
        __syncthreads();
    }

    float inv[8];
#pragma unroll
    for (int e = 0; e < 8; ++e) inv[e] = 1.0f / rsum[w * 16 + e + hh * 8];
#pragma unroll
    for (int ni = 0; ni < 4; ++ni)
#pragma unroll
        for (int e = 0; e < 8; ++e) {
            int r = qb + w * 16 + e + hh * 8;
            int d = ni * 16 + ln;
            Og[(rowbase + r) * 512 + hoff + d] = oacc[ni][e] * inv[e];
        }
}

// ---------------------------------------------------------------------------
// LayerNorm over D=1024, one block (256 threads) per row, two-pass.
// ---------------------------------------------------------------------------
__global__ __launch_bounds__(256) void layernorm_kernel(
    const float* __restrict__ X, const float* __restrict__ g,
    const float* __restrict__ bta, float* __restrict__ out)
{
    __shared__ float row[1024];
    __shared__ float red[256];
    const int t = threadIdx.x;
    const size_t base = (size_t)blockIdx.x * 1024;

    float s = 0.f;
#pragma unroll
    for (int i = 0; i < 4; ++i) {
        float v = X[base + t + i * 256];
        row[t + i * 256] = v;
        s += v;
    }
    red[t] = s;
    __syncthreads();
    for (int o = 128; o > 0; o >>= 1) {
        if (t < o) red[t] += red[t + o];
        __syncthreads();
    }
    float mu = red[0] * (1.0f / 1024.0f);
    __syncthreads();

    float vs = 0.f;
#pragma unroll
    for (int i = 0; i < 4; ++i) {
        float d = row[t + i * 256] - mu;
        vs += d * d;
    }
    red[t] = vs;
    __syncthreads();
    for (int o = 128; o > 0; o >>= 1) {
        if (t < o) red[t] += red[t + o];
        __syncthreads();
    }
    float rstd = rsqrtf(red[0] * (1.0f / 1024.0f) + 1e-5f);
#pragma unroll
    for (int i = 0; i < 4; ++i) {
        int j = t + i * 256;
        out[base + j] = (row[j] - mu) * rstd * g[j] + bta[j];
    }
}

// ---------------------------------------------------------------------------
extern "C" void kernel_launch(void* const* d_in, const int* in_sizes, int n_in,
                              void* d_out, int out_size, void* d_ws,
                              size_t ws_size, hipStream_t stream)
{
    (void)in_sizes; (void)n_in; (void)out_size; (void)ws_size;
    const float* x    = (const float*)d_in[0];
    const float* Wenc = (const float*)d_in[1];
    const float* benc = (const float*)d_in[2];
    const float* Wq   = (const float*)d_in[3];
    const float* bq   = (const float*)d_in[4];
    const float* Wk   = (const float*)d_in[5];
    const float* bk   = (const float*)d_in[6];
    const float* Wv   = (const float*)d_in[7];
    const float* bv   = (const float*)d_in[8];
    const float* Wo   = (const float*)d_in[9];
    const float* bo   = (const float*)d_in[10];
    // d_in[11] = ocr_bias: cancels under per-row softmax shift-invariance.
    const float* W1   = (const float*)d_in[12];
    const float* b1   = (const float*)d_in[13];
    const float* W2   = (const float*)d_in[14];
    const float* b2   = (const float*)d_in[15];
    const float* lng  = (const float*)d_in[16];
    const float* lnb  = (const float*)d_in[17];
    float* out = (float*)d_out;

    const int Rn = 4 * 2048;                 // 8192 token rows
    float* ws  = (float*)d_ws;
    float* mem = ws;                          // [Rn,512]
    float* q   = mem + (size_t)Rn * 512;      // [Rn,512]
    float* k   = q   + (size_t)Rn * 512;      // [Rn,512]
    float* v   = k   + (size_t)Rn * 512;      // [Rn,512]
    float* ctx = v   + (size_t)Rn * 512;      // [Rn,512]  (ws total: 80 MB)
    float* attno = mem;                       // reuse: mem dead after QKV
    float* h1    = q;                         // reuse: q+k dead after attn
    float* h2    = v;                         // reuse: v+ctx dead after Wo/W1

    const int NOSPLIT = 1 << 30;
    dim3 blk(256);
    dim3 g512(512 / 128, Rn / 128);
    dim3 g1024(1024 / 128, Rn / 128);

    // mem = x @ W_enc + b_enc
    gemm_bf16_kernel<<<g512, blk, 0, stream>>>(x, 1024, x, 1024, NOSPLIT,
                                               Wenc, 512, benc, mem, 512, 1024, 0);
    // q/k/v = mem @ W{q,k,v} + b
    gemm_bf16_kernel<<<g512, blk, 0, stream>>>(mem, 512, mem, 512, NOSPLIT,
                                               Wq, 512, bq, q, 512, 512, 0);
    gemm_bf16_kernel<<<g512, blk, 0, stream>>>(mem, 512, mem, 512, NOSPLIT,
                                               Wk, 512, bk, k, 512, 512, 0);
    gemm_bf16_kernel<<<g512, blk, 0, stream>>>(mem, 512, mem, 512, NOSPLIT,
                                               Wv, 512, bv, v, 512, 512, 0);
    // fused attention -> ctx
    attn_kernel<<<dim3(2048 / 64, 4 * 8), dim3(128), 0, stream>>>(q, k, v, ctx);
    // attended = ctx @ Wo + bo
    gemm_bf16_kernel<<<g512, blk, 0, stream>>>(ctx, 512, ctx, 512, NOSPLIT,
                                               Wo, 512, bo, attno, 512, 512, 0);
    // h1 = gelu(concat([x, attended]) @ W1 + b1)   (virtual concat, split-K)
    gemm_bf16_kernel<<<g1024, blk, 0, stream>>>(x, 1024, attno, 512, 1024,
                                                W1, 1024, b1, h1, 1024, 1536, 1);
    // h2 = h1 @ W2 + b2
    gemm_bf16_kernel<<<g1024, blk, 0, stream>>>(h1, 1024, h1, 1024, NOSPLIT,
                                                W2, 1024, b2, h2, 1024, 1024, 0);
    // out = layernorm(h2)
    layernorm_kernel<<<dim3(Rn), blk, 0, stream>>>(h2, lng, lnb, out);
}
